// OBM_GATv2Conv_68667937128572
// MI455X (gfx1250) — matrix operand order repaired and checked by reading, not compiled, not run on hardware
//
#include <hip/hip_runtime.h>
#include <hip/hip_bf16.h>
#include <math.h>

// Problem constants (from the reference)
#define NN      50000
#define EE      800000
#define FIN     64
#define HH      128
#define EFD     8
#define GG      10
#define GFDD    2
#define NEG_SLOPE 0.2f
#define MT      5            // M-tiles per wave: 3125 = 5 * 625, exact -> no guards

typedef __attribute__((ext_vector_type(16))) _Float16 v16h;
typedef __attribute__((ext_vector_type(8)))  float    v8f;

// ---------------------------------------------------------------------------
// Async copy one f32 from global to LDS (GLOBAL_LOAD_ASYNC_TO_LDS_B32,
// ASYNCcnt-tracked). Generic LDS pointer low 32 bits = LDS byte offset
// (flat-aperture rule).
// ---------------------------------------------------------------------------
__device__ __forceinline__ void async_copy_f32(float* ldsDst, const float* gSrc) {
    unsigned lds_off = (unsigned)(uintptr_t)ldsDst;
    unsigned long long ga = (unsigned long long)(uintptr_t)gSrc;
    asm volatile("global_load_async_to_lds_b32 %0, %1, off"
                 :: "v"(lds_off), "v"(ga) : "memory");
}
__device__ __forceinline__ void async_wait_all() {
    asm volatile("s_wait_asynccnt 0x0" ::: "memory");
}

// ---------------------------------------------------------------------------
// WMMA dense linear: Y[NN,HH] = X[NN,K] @ W[K,HH] + b
// Block = 8 waves; wave w owns N-tile tn=w (8*16 = HH columns) and computes an
// 80x16 strip (MT=5 M-tiles, 3125 = 5*625 exact) so the column-strided B
// fragment of W is loaded once per k-chunk and reused by 5 v_wmma ops, with no
// branches anywhere (EXEC all-ones at every WMMA, per ISA requirement).
// A layout (16x32 f16): lanes 0-15 row M=lane, elems j = K=k0+j;
//   lanes 16-31 same rows, elems j = K=k0+16+j.
// B layout (32x16 f16): lanes 0-15 col N=lane, elems j = K=k0+j;
//   lanes 16-31 col N=lane-16, elems j = K=k0+16+j.
// C/D: VGPR r -> (M = tile*16 + r + (lane>=16 ? 8:0), N = tn*16 + (lane&15)).
// ---------------------------------------------------------------------------
__global__ void gatv2_wmma_linear(const float* __restrict__ X,
                                  const float* __restrict__ W,
                                  const float* __restrict__ bvec,
                                  float* __restrict__ Y, int K) {
    const int lane = threadIdx.x & 31;
    const int tn   = threadIdx.x >> 5;        // 0..7 : N-tile
    const int mg   = blockIdx.x;              // M-group of MT tiles

    const int col = tn * 16 + (lane & 15);
    const int hiA = (lane & 16) ? 16 : 0;
    const int hiB = (lane & 16) ? 16 : 0;

    v8f c[MT];
#pragma unroll
    for (int i = 0; i < MT; ++i) c[i] = {};

    for (int k0 = 0; k0 < K; k0 += 32) {
        v16h b;
        const float* wp = W + (size_t)(k0 + hiB) * HH + col;
#pragma unroll
        for (int j = 0; j < 16; ++j) b[j] = (_Float16)wp[(size_t)j * HH];
#pragma unroll
        for (int i = 0; i < MT; ++i) {
            const int tm = mg * MT + i;
            v16h a;
            const float* xp = X + (size_t)(tm * 16 + (lane & 15)) * K + k0 + hiA;
#pragma unroll
            for (int j = 0; j < 8; ++j) a[j]     = (_Float16)xp[j];
#pragma unroll
            for (int j = 0; j < 8; ++j) a[8 + j] = (_Float16)xp[8 + j];
            // (neg_a, A, neg_b, B, c_mod, C, reuse_a, reuse_b)
            c[i] = __builtin_amdgcn_wmma_f32_16x16x32_f16(false, a, false, b,
                                                          (short)0, c[i],
                                                          false, false);
        }
    }

    const float bias = bvec[col];
#pragma unroll
    for (int i = 0; i < MT; ++i) {
        const int mbase = (mg * MT + i) * 16 + ((lane & 16) ? 8 : 0);
#pragma unroll
        for (int r = 0; r < 8; ++r)
            Y[(size_t)(mbase + r) * HH + col] = c[i][r] + bias;
    }
}

// ---------------------------------------------------------------------------
// Monotonic-int-encoded float atomic max (lowers to global_atomic_max/min).
// ---------------------------------------------------------------------------
__device__ __forceinline__ void atomicMaxFloat(float* addr, float val) {
    if (val >= 0.0f)
        atomicMax((int*)addr, __float_as_int(val));
    else
        atomicMin((unsigned int*)addr, __float_as_uint(val));
}

// ---------------------------------------------------------------------------
// Per-layer init: mx = -inf, den = 0
// ---------------------------------------------------------------------------
__global__ void gatv2_init_nodes(float* __restrict__ mx, float* __restrict__ den) {
    int i = blockIdx.x * blockDim.x + threadIdx.x;
    if (i < NN) { mx[i] = -INFINITY; den[i] = 0.0f; }
}

__global__ void gatv2_zero(float* __restrict__ p, int n) {
    int i = blockIdx.x * blockDim.x + threadIdx.x;
    if (i < n) p[i] = 0.0f;
}

// ---------------------------------------------------------------------------
// Edge pass 1: logits[e] = leaky_relu(xl[src] + xr[dst] + edge_attr@We) . att
// One wave per edge; lane owns 4 channels of H=128. We (8x128, 4KB) and att
// staged in LDS via GLOBAL_LOAD_ASYNC_TO_LDS_B32 + s_wait_asynccnt.
// xl/xr are L2-resident (25.6 MB each on a 192 MB L2) so the random row
// gathers are L2 traffic, not HBM.
// ---------------------------------------------------------------------------
__global__ void gatv2_edge_logits(const float* __restrict__ xl,
                                  const float* __restrict__ xr,
                                  const float* __restrict__ eattr,
                                  const int*   __restrict__ srcIdx,
                                  const int*   __restrict__ dstIdx,
                                  const float* __restrict__ We,
                                  const float* __restrict__ att,
                                  float* __restrict__ logits,
                                  float* __restrict__ mx) {
    __shared__ float sWe[EFD * HH];
    __shared__ float sAtt[HH];
    for (int i = threadIdx.x; i < EFD * HH; i += blockDim.x)
        async_copy_f32(&sWe[i], &We[i]);
    for (int i = threadIdx.x; i < HH; i += blockDim.x)
        async_copy_f32(&sAtt[i], &att[i]);
    async_wait_all();
    __syncthreads();

    const int lane = threadIdx.x & 31;
    const int wid  = (blockIdx.x * blockDim.x + threadIdx.x) >> 5;
    const int nw   = (gridDim.x * blockDim.x) >> 5;

    for (int e = wid; e < EE; e += nw) {
        const int s = srcIdx[e];
        const int d = dstIdx[e];
        const float* ap = xl + (size_t)s * HH + lane * 4;
        const float* bp = xr + (size_t)d * HH + lane * 4;
        float ea[EFD];
#pragma unroll
        for (int f = 0; f < EFD; ++f) ea[f] = eattr[(size_t)e * EFD + f];
        float partial = 0.0f;
#pragma unroll
        for (int j = 0; j < 4; ++j) {
            const int h = lane * 4 + j;
            float ev = 0.0f;
#pragma unroll
            for (int f = 0; f < EFD; ++f) ev = fmaf(ea[f], sWe[f * HH + h], ev);
            float v = ap[j] + bp[j] + ev;
            v = (v > 0.0f) ? v : NEG_SLOPE * v;
            partial = fmaf(v, sAtt[h], partial);
        }
#pragma unroll
        for (int off = 16; off > 0; off >>= 1)
            partial += __shfl_down(partial, off, 32);
        if (lane == 0) {
            logits[e] = partial;
            atomicMaxFloat(&mx[d], partial);
        }
    }
}

// ---------------------------------------------------------------------------
// Edge pass 2: ex[e] = exp(logit - mx[dst]); den[dst] += ex  (one thread/edge)
// ---------------------------------------------------------------------------
__global__ void gatv2_edge_exp(const float* __restrict__ logits,
                               const int*   __restrict__ dstIdx,
                               const float* __restrict__ mx,
                               float* __restrict__ exb,
                               float* __restrict__ den) {
    int e = blockIdx.x * blockDim.x + threadIdx.x;
    if (e < EE) {
        const int d = dstIdx[e];
        const float ex_ = __expf(logits[e] - mx[d]);
        exb[e] = ex_;
        atomicAdd(&den[d], ex_);
    }
}

// ---------------------------------------------------------------------------
// Edge pass 3: acc[dst] += (ex/den) * xl[src]  (wave per edge, 4 ch/lane)
// global_prefetch_b8 on the streamed per-edge arrays for the next iteration.
// ---------------------------------------------------------------------------
__global__ void gatv2_edge_scatter(const float* __restrict__ xl,
                                   const int*   __restrict__ srcIdx,
                                   const int*   __restrict__ dstIdx,
                                   const float* __restrict__ exb,
                                   const float* __restrict__ den,
                                   float* __restrict__ acc) {
    const int lane = threadIdx.x & 31;
    const int wid  = (blockIdx.x * blockDim.x + threadIdx.x) >> 5;
    const int nw   = (gridDim.x * blockDim.x) >> 5;
    for (int e = wid; e < EE; e += nw) {
        const int en = e + nw;
        if (en < EE) {                       // prefetch next iteration's streams
            __builtin_prefetch(&srcIdx[en], 0, 1);
            __builtin_prefetch(&dstIdx[en], 0, 1);
            __builtin_prefetch(&exb[en], 0, 1);
        }
        const int s = srcIdx[e];
        const int d = dstIdx[e];
        const float alpha = exb[e] / fmaxf(den[d], 1e-16f);
        const float* ap = xl + (size_t)s * HH + lane * 4;
        float*       op = acc + (size_t)d * HH + lane * 4;
#pragma unroll
        for (int j = 0; j < 4; ++j)
            atomicAdd(&op[j], alpha * ap[j]);
    }
}

// ---------------------------------------------------------------------------
// Epilogue: x_next = relu(acc + bias)
// ---------------------------------------------------------------------------
__global__ void gatv2_epilogue(const float* __restrict__ acc,
                               const float* __restrict__ bias,
                               float* __restrict__ out) {
    int i = blockIdx.x * blockDim.x + threadIdx.x;
    if (i < NN * HH) {
        const float v = acc[i] + bias[i & (HH - 1)];
        out[i] = v > 0.0f ? v : 0.0f;
    }
}

// ---------------------------------------------------------------------------
// Head: out[n] = x3[n,:] . hW[0:128] + gf0*hW[128] + gf1*hW[129] + hb
// Torch reshape chain collapsed analytically: with a = n/5000,
//   gf_b(n) = graph_features[ 2*(a%5)+b ][ a<5 ? 0 : 1 ]
// ---------------------------------------------------------------------------
__global__ void gatv2_head(const float* __restrict__ x3,
                           const float* __restrict__ gfp,
                           const float* __restrict__ hW,
                           const float* __restrict__ hb,
                           float* __restrict__ out) {
    const int lane = threadIdx.x & 31;
    const int wid  = (blockIdx.x * blockDim.x + threadIdx.x) >> 5;
    const int nw   = (gridDim.x * blockDim.x) >> 5;
    for (int n = wid; n < NN; n += nw) {
        const float* row = x3 + (size_t)n * HH;
        float partial = 0.0f;
#pragma unroll
        for (int j = 0; j < 4; ++j) {
            const int h = lane + j * 32;
            partial = fmaf(row[h], hW[h], partial);
        }
#pragma unroll
        for (int off = 16; off > 0; off >>= 1)
            partial += __shfl_down(partial, off, 32);
        if (lane == 0) {
            const int a    = n / (NN / GG);     // n / 5000
            const int dsel = (a < 5) ? 0 : 1;
            const int am   = a % 5;
            const float g0 = gfp[(2 * am + 0) * GFDD + dsel];
            const float g1 = gfp[(2 * am + 1) * GFDD + dsel];
            out[n] = partial + g0 * hW[HH] + g1 * hW[HH + 1] + hb[0];
        }
    }
}

// ---------------------------------------------------------------------------
extern "C" void kernel_launch(void* const* d_in, const int* in_sizes, int n_in,
                              void* d_out, int out_size, void* d_ws, size_t ws_size,
                              hipStream_t stream) {
    (void)in_sizes; (void)n_in; (void)out_size; (void)ws_size;

    const float* x      = (const float*)d_in[0];
    const int*   eidx   = (const int*)  d_in[1];
    const float* eattr  = (const float*)d_in[2];
    const float* gfp    = (const float*)d_in[3];
    // d_in[4] = num_graphs (compile-time GG)
    const float* Wl[3]   = {(const float*)d_in[5],  (const float*)d_in[12], (const float*)d_in[19]};
    const float* bl[3]   = {(const float*)d_in[6],  (const float*)d_in[13], (const float*)d_in[20]};
    const float* Wr[3]   = {(const float*)d_in[7],  (const float*)d_in[14], (const float*)d_in[21]};
    const float* br[3]   = {(const float*)d_in[8],  (const float*)d_in[15], (const float*)d_in[22]};
    const float* We[3]   = {(const float*)d_in[9],  (const float*)d_in[16], (const float*)d_in[23]};
    const float* att[3]  = {(const float*)d_in[10], (const float*)d_in[17], (const float*)d_in[24]};
    const float* bias[3] = {(const float*)d_in[11], (const float*)d_in[18], (const float*)d_in[25]};
    const float* hW = (const float*)d_in[26];
    const float* hb = (const float*)d_in[27];

    const int* srcIdx = eidx;        // edge_index[0]
    const int* dstIdx = eidx + EE;   // edge_index[1]

    // Workspace layout (floats): xl, xr, acc, bufX (NN*HH each), logits, ex (EE), mx, den (NN)
    float* ws     = (float*)d_ws;
    float* xl     = ws;
    float* xrr    = xl   + (size_t)NN * HH;
    float* acc    = xrr  + (size_t)NN * HH;
    float* bufX   = acc  + (size_t)NN * HH;
    float* logits = bufX + (size_t)NN * HH;
    float* exb    = logits + (size_t)EE;
    float* mx     = exb    + (size_t)EE;
    float* den    = mx     + (size_t)NN;

    const int TILES_M     = NN / 16;                 // 3125
    const int GEMM_BLOCKS = TILES_M / MT;            // 625, exact
    const int NODE_BLOCKS = (NN + 255) / 256;
    const int NH_BLOCKS   = (NN * HH + 255) / 256;
    const int E_BLOCKS    = (EE + 255) / 256;
    const int WAVE_BLOCKS = 2048;                    // grid-stride wave loops
    const int HEAD_BLOCKS = NN / 8;                  // 8 waves per block, exact

    for (int l = 0; l < 3; ++l) {
        const float* xin = (l == 0) ? x : bufX;
        const int K = (l == 0) ? FIN : HH;

        gatv2_wmma_linear<<<GEMM_BLOCKS, 256, 0, stream>>>(xin, Wl[l], bl[l], xl,  K);
        gatv2_wmma_linear<<<GEMM_BLOCKS, 256, 0, stream>>>(xin, Wr[l], br[l], xrr, K);
        gatv2_init_nodes<<<NODE_BLOCKS, 256, 0, stream>>>(mx, den);
        gatv2_zero<<<NH_BLOCKS, 256, 0, stream>>>(acc, NN * HH);
        gatv2_edge_logits<<<WAVE_BLOCKS, 256, 0, stream>>>(xl, xrr, eattr, srcIdx, dstIdx,
                                                           We[l], att[l], logits, mx);
        gatv2_edge_exp<<<E_BLOCKS, 256, 0, stream>>>(logits, dstIdx, mx, exb, den);
        gatv2_edge_scatter<<<WAVE_BLOCKS, 256, 0, stream>>>(xl, srcIdx, dstIdx, exb, den, acc);
        gatv2_epilogue<<<NH_BLOCKS, 256, 0, stream>>>(acc, bias[l], bufX);
    }
    gatv2_head<<<HEAD_BLOCKS, 256, 0, stream>>>(bufX, gfp, hW, hb, (float*)d_out);
}
